// CppnPotentialCAStep_10917806867136
// MI455X (gfx1250) — compile-verified, matrix operand order checked by reading, not run on hardware
//
#include <hip/hip_runtime.h>

typedef __attribute__((ext_vector_type(16))) _Float16 v16h;
typedef __attribute__((ext_vector_type(8)))  float    v8f;
typedef __attribute__((ext_vector_type(4)))  unsigned int v4u;

#define S 96
#define C 16
#define NK 32
#define KS 15
#define XPAD 112            // padded row length (f16 elems): 110 used + 2 spare for b128 tails
#define LDS_ROW 80          // LDS row stride in bytes (64B data + 16B pad, keeps 16B align)
#define LDS_CH  (30*LDS_ROW)
#define LDS_BYTES (16*LDS_CH)   // 38400
#define NFRAG 512           // f16 elems per A fragment (32 lanes * 16)

// ---------------------------------------------------------------------------
// Prep 1: channel-major, x-halo-padded f16 copy of the input volume.
// chpad[c][z][y][xp] = (f16) x[z][y][(xp-7) mod 96][c]
// ---------------------------------------------------------------------------
__global__ void pack_kernel(const float* __restrict__ x, _Float16* __restrict__ chpad) {
    int tid = blockIdx.x * blockDim.x + threadIdx.x;
    if (tid >= C * S * S * XPAD) return;
    int xp = tid % XPAD;
    int r  = tid / XPAD;
    int y  = r % S; r /= S;
    int z  = r % S;
    int c  = r / S;
    int xs = (xp + (S - 7)) % S;                // wrap padding
    chpad[tid] = (_Float16)x[((z * S + y) * S + xs) * C + c];
}

// ---------------------------------------------------------------------------
// Prep 2: banded weight fragments pre-packed in the WMMA f16 A-operand
// register layout.  A[M,K] = w[k,dz,dy,K-M] for 0 <= K-M < 15, else 0.
// Layout (16-bit A 16x32, wave32):
//   lanes 0-15 : M=lane,    halves v0..v3 = K 0..7,  v4..v7 = K 16..23
//   lanes 16-31: M=lane-16, halves v0..v3 = K 8..15, v4..v7 = K 24..31
// Stored per fragment as 32 lanes x 16 f16 (32B/lane) so the hot loop loads
// a lane's operand with two global_load_b128.
// ---------------------------------------------------------------------------
__global__ void afrag_kernel(const float* __restrict__ kern, _Float16* __restrict__ afrag) {
    int tid = blockIdx.x * blockDim.x + threadIdx.x;
    if (tid >= NK * KS * KS * 32) return;
    int lane = tid & 31;
    int f    = tid >> 5;                         // fragment index = (k*15+dz)*15+dy
    int dy = f % KS; int t = f / KS;
    int dz = t % KS; int k = t / KS;
    int M  = lane & 15;
    const float* kb = kern + ((k * KS + dz) * KS + dy) * KS;
    _Float16* ob = afrag + (size_t)f * NFRAG + lane * 16;
#pragma unroll
    for (int idx = 0; idx < 16; ++idx) {
        int v = idx >> 1, h = idx & 1;
        int K;
        if (lane < 16) K = (v < 4) ? (2 * v + h)      : (16 + 2 * (v - 4) + h);
        else           K = (v < 4) ? (8 + 2 * v + h)  : (24 + 2 * (v - 4) + h);
        int dx = K - M;
        float w = (dx >= 0 && dx < KS) ? kb[dx] : 0.0f;
        ob[idx] = (_Float16)w;
    }
}

// ---------------------------------------------------------------------------
// Main: one block = one 16(x) x 16(y) x 1(z) output tile, all 32 kernels.
// 8 waves x 4 kernels each; per dz: stage rows to LDS, then 4x15 WMMAs/wave.
// ---------------------------------------------------------------------------
__global__ __launch_bounds__(256) void ca_main(
    const _Float16* __restrict__ chpad, const _Float16* __restrict__ afrag,
    const float* __restrict__ xin, const float* __restrict__ mArr,
    const float* __restrict__ sArr, const float* __restrict__ Tptr,
    const int* __restrict__ c0Arr, const int* __restrict__ c1Arr,
    float* __restrict__ outp)
{
    __shared__ __align__(16) unsigned char smem[LDS_BYTES];
    __shared__ int c1s[NK];

    const int tid  = threadIdx.x;
    const int lane = tid & 31;
    const int wave = tid >> 5;
    const int x0 = blockIdx.x * 16;
    const int y0 = blockIdx.y * 16;
    const int z0 = blockIdx.z;

    if (tid < NK) c1s[tid] = c1Arr[tid];

    const int kbase = wave * 4;
    float mk[4], sk[4];
    int   c0k[4];
#pragma unroll
    for (int i = 0; i < 4; ++i) {
        mk[i]  = mArr[kbase + i];
        sk[i]  = sArr[kbase + i];
        c0k[i] = c0Arr[kbase + i];
    }

    const v8f vzero = {0.f, 0.f, 0.f, 0.f, 0.f, 0.f, 0.f, 0.f};
    v8f acc[4];
#pragma unroll
    for (int i = 0; i < 4; ++i) acc[i] = vzero;

    const int n = lane & 15;                      // output column (N = y index)
    const int k0b = (lane < 16) ? 0 : 16;         // byte offset of this lane's first K chunk

    for (int dz = 0; dz < KS; ++dz) {
        const int zz = (z0 + dz + (S - 7)) % S;

        // ---- stage 16 channels x 30 rows x 32 f16 into LDS (b128 chunks) ----
        for (int t = tid; t < 16 * 30 * 4; t += 256) {
            int part = t & 3;
            int row  = t >> 2;
            int c = row / 30;
            int j = row - c * 30;
            int yy = (y0 + j + (S - 7)) % S;
            const v4u* g = (const v4u*)(chpad + (((c * S + zz) * S + yy) * XPAD + x0)) + part;
            *(v4u*)(smem + row * LDS_ROW + part * 16) = *g;
        }
        __syncthreads();

        const _Float16* ab[4];
#pragma unroll
        for (int i = 0; i < 4; ++i)
            ab[i] = afrag + (size_t)(((kbase + i) * KS + dz) * KS) * NFRAG + lane * 16;

        for (int dy = 0; dy < KS; ++dy) {
            const int jrow = n + dy;
#pragma unroll
            for (int i = 0; i < 4; ++i) {
                v16h a = *(const v16h*)(ab[i] + dy * NFRAG);
                const unsigned char* lb = smem + c0k[i] * LDS_CH + jrow * LDS_ROW + k0b;
                union { v4u u[2]; v16h h; } bu;
                bu.u[0] = *(const v4u*)(lb);
                bu.u[1] = *(const v4u*)(lb + 32);
                acc[i] = __builtin_amdgcn_wmma_f32_16x16x32_f16(
                    false, a, false, bu.h, (short)0, acc[i], false, false);
            }
        }
        __syncthreads();
    }

    // ---- growth, written to LDS per-kernel for a deterministic segment-sum ----
    float* gbuf = (float*)smem;                   // 32 kernels x 16 x 16 = 32KB
    const int mhi = (lane >> 4) << 3;             // C/D layout: M = r + 8*(lane/16)
#pragma unroll
    for (int i = 0; i < 4; ++i) {
        const float inv2s2 = 0.5f / (sk[i] * sk[i]);
#pragma unroll
        for (int r = 0; r < 8; ++r) {
            float d = acc[i][r] - mk[i];
            float g = __expf(-(d * d) * inv2s2) * 2.0f - 1.0f;
            gbuf[(kbase + i) * 256 + (r + mhi) * 16 + n] = g;
        }
    }
    __syncthreads();

    // ---- deterministic segment-sum + fused add/clip/store ----
    const float invT = 1.0f / Tptr[0];
    const int ny = tid >> 4;                      // y within tile
    const int mx = tid & 15;                      // x within tile
    const size_t sb = ((((size_t)z0) * S + (y0 + ny)) * S + (x0 + mx)) * C;
#pragma unroll
    for (int c = 0; c < C; ++c) {
        float f = 0.0f;
#pragma unroll
        for (int k = 0; k < NK; ++k)
            f += (c1s[k] == c) ? gbuf[k * 256 + mx * 16 + ny] : 0.0f;
        float v = xin[sb + c] + f * invT;
        v = fminf(fmaxf(v, 0.0f), 10.0f);
        outp[sb + c] = v;
    }
}

// ---------------------------------------------------------------------------
extern "C" void kernel_launch(void* const* d_in, const int* in_sizes, int n_in,
                              void* d_out, int out_size, void* d_ws, size_t ws_size,
                              hipStream_t stream) {
    (void)in_sizes; (void)n_in; (void)out_size; (void)ws_size;

    const float* x    = (const float*)d_in[0];
    const float* kern = (const float*)d_in[1];
    const float* mA   = (const float*)d_in[2];
    const float* sA   = (const float*)d_in[3];
    const float* T    = (const float*)d_in[4];
    const int*   c0   = (const int*)d_in[5];
    const int*   c1   = (const int*)d_in[6];
    float* out = (float*)d_out;

    // workspace: 33.0 MB padded f16 volume + 7.4 MB pre-packed A fragments
    _Float16* chpad = (_Float16*)d_ws;
    _Float16* afrag = (_Float16*)((char*)d_ws + (size_t)C * S * S * XPAD * 2);

    const int n1 = C * S * S * XPAD;
    pack_kernel<<<(n1 + 255) / 256, 256, 0, stream>>>(x, chpad);

    const int n2 = NK * KS * KS * 32;
    afrag_kernel<<<(n2 + 255) / 256, 256, 0, stream>>>(kern, afrag);

    dim3 grid(S / 16, S / 16, S);
    ca_main<<<grid, 256, 0, stream>>>(chpad, afrag, x, mA, sA, T, c0, c1, out);
}
